// MultiHeadAttention_43825846288911
// MI455X (gfx1250) — matrix operand order repaired and checked by reading, not compiled, not run on hardware
//
#include <hip/hip_runtime.h>
#include <math.h>
#include <stdint.h>

// ---------------------------------------------------------------------------
// Multi-head attention forward for MI455X (gfx1250), bf16 WMMA everywhere.
// B=2, S=2048, E=768, H=12, DK=64, scale = 1/sqrt(64) = 0.125
// ---------------------------------------------------------------------------

typedef __attribute__((ext_vector_type(16))) __bf16 v16bf;
typedef __attribute__((ext_vector_type(8)))  __bf16 bf16x8;
typedef __attribute__((ext_vector_type(4)))  __bf16 bf16x4;
typedef __attribute__((ext_vector_type(8)))  float  v8f;
typedef __attribute__((ext_vector_type(4)))  float  f32x4;

#define BATCH 2
#define SEQ   2048
#define EMB   768
#define NH    12
#define HD    64
#define NTOK  (BATCH * SEQ)          // 4096 rows in the token dimension

// D = A*B + C, bf16 inputs, f32 accumulate (V_WMMA_F32_16X16X32_BF16)
static __device__ __forceinline__ v8f wmma_bf16(v16bf a, v16bf b, v8f c) {
  return __builtin_amdgcn_wmma_f32_16x16x32_bf16(
      /*neg_a=*/false, a, /*neg_b=*/false, b,
      /*c_mod=*/(short)0, c, /*reuse_a=*/false, /*reuse_b=*/false);
}

// A-matrix fragment (16x32 bf16, MxK). Layout used here:
//  lane = m + 16*half holds: k = 16*half+e (contiguous)
static __device__ __forceinline__ v16bf afrag(const __bf16* row, int half) {
  bf16x8 lo = *(const bf16x8*)(row + 16 * half);
  bf16x8 hi = *(const bf16x8*)(row + 16 * half + 8);
  v16bf r;
#pragma unroll
  for (int i = 0; i < 8; ++i) { r[i] = lo[i]; r[i + 8] = hi[i]; }
  return r;
}

// ---------------------------------------------------------------------------
// fp32 -> bf16 conversion (x4 vectorized)
// ---------------------------------------------------------------------------
__global__ void cvt_f32_to_bf16(const float* __restrict__ src,
                                __bf16* __restrict__ dst, int n4) {
  int i = blockIdx.x * blockDim.x + threadIdx.x;
  if (i >= n4) return;
  f32x4 v = ((const f32x4*)src)[i];
  bf16x4 o;
#pragma unroll
  for (int j = 0; j < 4; ++j) o[j] = (__bf16)v[j];
  ((bf16x4*)dst)[i] = o;
}

// ---------------------------------------------------------------------------
// GEMM: Y = X * W^T + bias.  X: [NTOK, EMB] bf16 row-major. W: [out,in] bf16.
// Block = 128 threads (4 waves) -> 128x64 output tile; wave -> 32x64.
// B tile (64 cols x 32 k, 4KB) is staged in LDS via global_load_async_to_lds,
// double-buffered and tracked with ASYNCcnt; all 4 waves share it.
//   mode: 0 = write bf16 [B,H,S,HD] (Q,K)
//         1 = write bf16 [B,H,HD,S] (V transposed)
//         2 = write f32  [NTOK, EMB] (final output)
// ---------------------------------------------------------------------------
__global__ void gemm_xwT(const __bf16* __restrict__ X,
                         const __bf16* __restrict__ W,
                         const float* __restrict__ bias,
                         void* __restrict__ out, int mode) {
  __shared__ __attribute__((aligned(16))) __bf16 Bt[2][64 * 32];  // [buf][col*32+k]
  const int tid = threadIdx.x;
  const int lane = tid & 31;
  const int w = tid >> 5;
  const int mb = blockIdx.x & 31;   // 32 row blocks of 128
  const int nb = blockIdx.x >> 5;   // 12 col blocks of 64
  if (nb >= EMB / 64) return;
  const int n = lane & 15;
  const int half = lane >> 4;
  const int col0 = nb * 64;
  const int row0 = mb * 128 + w * 32;                 // this wave's 32 rows
  const __bf16* xrow0 = X + (size_t)(row0 + n) * EMB;
  const __bf16* xrow1 = X + (size_t)(row0 + 16 + n) * EMB;

  // Cooperative async copy: 64x32 bf16 tile = 256 x 16B chunks, 2 per thread.
  auto issueB = [&](int buf, int kk) {
#pragma unroll
    for (int c = 0; c < 2; ++c) {
      const int chunk = tid * 2 + c;          // 0..255
      const int col = chunk >> 2;             // 0..63
      const int part = chunk & 3;             // 16B piece of the 64B k-row
      const __bf16* gp = W + (size_t)(col0 + col) * EMB + kk + part * 8;
      const uint32_t lp = (uint32_t)(uintptr_t)(&Bt[buf][col * 32 + part * 8]);
      asm volatile("global_load_async_to_lds_b128 %0, %1, off"
                   :: "v"(lp), "v"((unsigned long long)(uintptr_t)gp)
                   : "memory");
    }
  };

  v8f c0[4] = {}, c1[4] = {};
  issueB(0, 0);
  for (int kk = 0; kk < EMB; kk += 32) {
    const int cur = (kk >> 5) & 1;
    if (kk + 32 < EMB) {
      issueB(cur ^ 1, kk + 32);
      asm volatile("s_wait_asynccnt 0x2" ::: "memory");   // cur buffer landed
    } else {
      asm volatile("s_wait_asynccnt 0x0" ::: "memory");
    }
    __syncthreads();                                      // all threads' chunks visible

    v16bf a0 = afrag(xrow0 + kk, half);
    v16bf a1 = afrag(xrow1 + kk, half);
#pragma unroll
    for (int nt = 0; nt < 4; ++nt) {
      // B fragment from LDS: lane holds column o = col0+nt*16+n,
      // contiguous k = 16*half + e -> 32B at Bt[cur][(nt*16+n)*32 + 16*half]
      const v16bf b = *(const v16bf*)&Bt[cur][(nt * 16 + n) * 32 + 16 * half];
      c0[nt] = wmma_bf16(a0, b, c0[nt]);
      c1[nt] = wmma_bf16(a1, b, c1[nt]);
    }
    __syncthreads();                       // protect cur buffer before overwrite
  }

  // Epilogue for the two 16-row groups
  auto store_tile = [&](v8f* c, int rowbase) {
#pragma unroll
    for (int nt = 0; nt < 4; ++nt) {
      const int o = col0 + nt * 16 + n;
      const float bv = bias[o];
#pragma unroll
      for (int r = 0; r < 8; ++r) {
        const int m = r + 8 * half;       // C layout: lane n+16*(m>=8), vgpr m&7
        const int g = rowbase + m;        // token index
        float v = c[nt][r] + bv;
        if (mode == 2) {
          ((float*)out)[(size_t)g * EMB + o] = v;
        } else {
          const int bi = g >> 11, s = g & (SEQ - 1);
          const int h = o >> 6, d = o & (HD - 1);
          size_t idx = (mode == 0)
              ? ((size_t)(bi * NH + h) * SEQ + s) * HD + d     // [B,H,S,HD]
              : ((size_t)(bi * NH + h) * HD + d) * SEQ + s;    // [B,H,HD,S]
          ((__bf16*)out)[idx] = (__bf16)v;
        }
      }
    }
  };
  store_tile(c0, row0);
  store_tile(c1, row0 + 16);
}

// ---------------------------------------------------------------------------
// Fused flash attention: one wave per 16-query tile of one (b,h).
// Q,K: [B,H,S,HD] bf16.  VT: [B,H,HD,S] bf16.  mask8: attn_mask bytes,
// key-padding read from attn_mask[b,0,0,k]; causal mask analytic.
// concat: [B,S,E] bf16.
// ---------------------------------------------------------------------------
__global__ void attn_fused(const __bf16* __restrict__ Q,
                           const __bf16* __restrict__ K,
                           const __bf16* __restrict__ VT,
                           const unsigned char* __restrict__ mask8,
                           __bf16* __restrict__ concat) {
  __shared__ float pbuf[4][16 * 32];          // per-wave P tile (fp32)
  const int lane = threadIdx.x & 31;
  const int w = threadIdx.x >> 5;
  const int wave = blockIdx.x * (blockDim.x >> 5) + w;
  const int qt = wave & 127;                  // 128 q-tiles per (b,h)
  const int bh = wave >> 7;
  if (bh >= BATCH * NH) return;
  const int b = bh / NH, h = bh % NH;
  const int n = lane & 15, half = lane >> 4;
  const int qbase = qt * 16;
  const unsigned char* pad = mask8 + (size_t)b * NH * SEQ * SEQ;  // [b,0,0,:]

  // Q fragments, constant over the key loop (d 0..31 and d 32..63)
  const __bf16* qrow = Q + ((size_t)bh * SEQ + qbase + n) * HD;
  const v16bf qa0 = afrag(qrow, half);
  const v16bf qa1 = afrag(qrow + 32, half);

  v8f acc[4] = {};
  float rmax[8], rsum[8];
#pragma unroll
  for (int r = 0; r < 8; ++r) { rmax[r] = -1e30f; rsum[r] = 0.f; }

  float* P = pbuf[w];

  // causal: keys beyond qbase+15 are fully masked -> loop bound
  for (int kb = 0; kb < qbase + 16; kb += 32) {
    // ---- scores for 32 keys (two 16-key tiles), 4 WMMAs ----
    const __bf16* krow0 = K + ((size_t)bh * SEQ + kb + n) * HD;
    const __bf16* krow1 = krow0 + 16 * HD;
    v8f s0 = {}, s1 = {};
    s0 = wmma_bf16(qa0, *(const v16bf*)(krow0 + 16 * half), s0);
    s0 = wmma_bf16(qa1, *(const v16bf*)(krow0 + 32 + 16 * half), s0);
    s1 = wmma_bf16(qa0, *(const v16bf*)(krow1 + 16 * half), s1);
    s1 = wmma_bf16(qa1, *(const v16bf*)(krow1 + 32 + 16 * half), s1);

    const int k0 = kb + n, k1 = kb + 16 + n;
    const bool pad0 = pad[k0] != 0;
    const bool pad1 = pad[k1] != 0;

    // ---- online softmax (rows live in 16-lane groups; xor 1,2,4,8) ----
#pragma unroll
    for (int r = 0; r < 8; ++r) {
      const int q = qbase + r + 8 * half;
      float e0 = (pad0 || (k0 > q)) ? -1e30f : s0[r] * 0.125f;
      float e1 = (pad1 || (k1 > q)) ? -1e30f : s1[r] * 0.125f;
      float bm = fmaxf(e0, e1);
      bm = fmaxf(bm, __shfl_xor(bm, 1));
      bm = fmaxf(bm, __shfl_xor(bm, 2));
      bm = fmaxf(bm, __shfl_xor(bm, 4));
      bm = fmaxf(bm, __shfl_xor(bm, 8));
      const float mnew = fmaxf(rmax[r], bm);
      const float corr = __expf(rmax[r] - mnew);
      const float p0 = __expf(e0 - mnew);
      const float p1 = __expf(e1 - mnew);
      float ps = p0 + p1;
      ps += __shfl_xor(ps, 1);
      ps += __shfl_xor(ps, 2);
      ps += __shfl_xor(ps, 4);
      ps += __shfl_xor(ps, 8);
      rsum[r] = rsum[r] * corr + ps;
      rmax[r] = mnew;
#pragma unroll
      for (int nt = 0; nt < 4; ++nt) acc[nt][r] *= corr;
      const int m = r + 8 * half;
      P[m * 32 + n] = p0;          // C-layout store of the P tile
      P[m * 32 + 16 + n] = p1;
    }

    // LDS RAW within the wave: DS ops are in-order, but be explicit.
    asm volatile("s_wait_dscnt 0x0" ::: "memory");

    // ---- reload P as a 16x32 bf16 A-fragment (transpose via LDS) ----
    v16bf pa;
    {
      const float* prow = P + n * 32;            // this lane's A row m = n
      f32x4 f0 = *(const f32x4*)(prow + 16 * half);
      f32x4 f1 = *(const f32x4*)(prow + 16 * half + 4);
      f32x4 f2 = *(const f32x4*)(prow + 16 * half + 8);
      f32x4 f3 = *(const f32x4*)(prow + 16 * half + 12);
#pragma unroll
      for (int i = 0; i < 4; ++i) {
        pa[i]      = (__bf16)f0[i];
        pa[i + 4]  = (__bf16)f1[i];
        pa[i + 8]  = (__bf16)f2[i];
        pa[i + 12] = (__bf16)f3[i];
      }
    }

    // ---- acc += P @ V : B-frags contiguous along keys thanks to V^T ----
#pragma unroll
    for (int nt = 0; nt < 4; ++nt) {
      const __bf16* vp =
          VT + ((size_t)bh * HD + nt * 16 + n) * SEQ + kb + 16 * half;
      acc[nt] = wmma_bf16(pa, *(const v16bf*)vp, acc[nt]);
    }
  }

  // ---- normalize and scatter into concat [B,S,E] ----
#pragma unroll
  for (int r = 0; r < 8; ++r) {
    const float inv = rsum[r] > 0.f ? 1.0f / rsum[r] : 0.f;
    const int m = r + 8 * half;
    const size_t g = (size_t)b * SEQ + qbase + m;
#pragma unroll
    for (int nt = 0; nt < 4; ++nt) {
      concat[g * EMB + h * HD + nt * 16 + n] = (__bf16)(acc[nt][r] * inv);
    }
  }
}

// ---------------------------------------------------------------------------
// Host-side orchestration (graph-capture safe: only async launches on stream)
// ---------------------------------------------------------------------------
extern "C" void kernel_launch(void* const* d_in, const int* in_sizes, int n_in,
                              void* d_out, int out_size, void* d_ws,
                              size_t ws_size, hipStream_t stream) {
  (void)in_sizes; (void)n_in; (void)out_size; (void)ws_size;
  const float* q_in = (const float*)d_in[0];
  const float* k_in = (const float*)d_in[1];
  const float* v_in = (const float*)d_in[2];
  const unsigned char* attn_mask = (const unsigned char*)d_in[3];
  // d_in[4] = dec_casual_mask: reconstructed analytically (k > q)
  const float* Wq = (const float*)d_in[5];  const float* bq = (const float*)d_in[6];
  const float* Wk = (const float*)d_in[7];  const float* bk = (const float*)d_in[8];
  const float* Wv = (const float*)d_in[9];  const float* bv = (const float*)d_in[10];
  const float* Wo = (const float*)d_in[11]; const float* bo = (const float*)d_in[12];

  // Workspace carve (256B aligned)
  char* ws = (char*)d_ws;
  size_t off = 0;
  auto carve = [&](size_t elems) -> __bf16* {
    __bf16* p = (__bf16*)(ws + off);
    off += ((elems * sizeof(__bf16) + 255) / 256) * 256;
    return p;
  };
  const size_t NX = (size_t)NTOK * EMB;       // 4096*768
  const size_t NW = (size_t)EMB * EMB;        // 768*768
  const size_t NQ = (size_t)BATCH * NH * SEQ * HD;
  __bf16* xq = carve(NX);
  __bf16* xk = carve(NX);
  __bf16* xv = carve(NX);
  __bf16* wq = carve(NW);
  __bf16* wk = carve(NW);
  __bf16* wv = carve(NW);
  __bf16* wo = carve(NW);
  __bf16* Qp = carve(NQ);
  __bf16* Kp = carve(NQ);
  __bf16* Vt = carve(NQ);                     // [B,H,HD,S]
  __bf16* cc = carve(NX);                     // concat [B,S,E]

  // 1) fp32 -> bf16
  {
    const int tb = 256;
    int n4x = (int)(NX / 4), n4w = (int)(NW / 4);
    hipLaunchKernelGGL(cvt_f32_to_bf16, dim3((n4x + tb - 1) / tb), dim3(tb), 0, stream, q_in, xq, n4x);
    hipLaunchKernelGGL(cvt_f32_to_bf16, dim3((n4x + tb - 1) / tb), dim3(tb), 0, stream, k_in, xk, n4x);
    hipLaunchKernelGGL(cvt_f32_to_bf16, dim3((n4x + tb - 1) / tb), dim3(tb), 0, stream, v_in, xv, n4x);
    hipLaunchKernelGGL(cvt_f32_to_bf16, dim3((n4w + tb - 1) / tb), dim3(tb), 0, stream, Wq, wq, n4w);
    hipLaunchKernelGGL(cvt_f32_to_bf16, dim3((n4w + tb - 1) / tb), dim3(tb), 0, stream, Wk, wk, n4w);
    hipLaunchKernelGGL(cvt_f32_to_bf16, dim3((n4w + tb - 1) / tb), dim3(tb), 0, stream, Wv, wv, n4w);
    hipLaunchKernelGGL(cvt_f32_to_bf16, dim3((n4w + tb - 1) / tb), dim3(tb), 0, stream, Wo, wo, n4w);
  }

  // 2) QKV projections: 32 row-blocks x 12 col-blocks = 384 blocks x 128 thr
  const dim3 ggrid(384), gblk(128);
  hipLaunchKernelGGL(gemm_xwT, ggrid, gblk, 0, stream, xq, wq, bq, (void*)Qp, 0);
  hipLaunchKernelGGL(gemm_xwT, ggrid, gblk, 0, stream, xk, wk, bk, (void*)Kp, 0);
  hipLaunchKernelGGL(gemm_xwT, ggrid, gblk, 0, stream, xv, wv, bv, (void*)Vt, 1);

  // 3) fused attention: 24 (b,h) x 128 q-tiles = 3072 waves
  hipLaunchKernelGGL(attn_fused, dim3(768), dim3(128), 0, stream, Qp, Kp, Vt, attn_mask, cc);

  // 4) output projection -> fp32 d_out
  hipLaunchKernelGGL(gemm_xwT, ggrid, gblk, 0, stream, cc, wo, bo, d_out, 2);
}
